// Attention_11364483465411
// MI455X (gfx1250) — matrix operand order repaired and checked by reading, not compiled, not run on hardware
//
#include <hip/hip_runtime.h>

// ---------------------------------------------------------------------------
// CDNA5 (gfx1250) wave32 WMMA attention pipeline with TDM tile staging.
// ---------------------------------------------------------------------------

typedef _Float16     h16x16 __attribute__((ext_vector_type(16)));
typedef _Float16     h16x8  __attribute__((ext_vector_type(8)));
typedef float        f32x8  __attribute__((ext_vector_type(8)));
typedef unsigned int u32x4  __attribute__((ext_vector_type(4)));
typedef int          i32x4  __attribute__((ext_vector_type(4)));
typedef int          i32x8  __attribute__((ext_vector_type(8)));

#define BATCH 2
#define SEQ   2048
#define DIM   1024
#define NHEAD 16
#define HDIM  64

__device__ __forceinline__ f32x8 wmma_f16(h16x16 a, h16x16 b, f32x8 c) {
    // v_wmma_f32_16x16x32_f16  (8-arg form, probe-confirmed)
    return __builtin_amdgcn_wmma_f32_16x16x32_f16(false, a, false, b,
                                                  (short)0, c, false, false);
}

// TDM: load a 32-row x 64-half tile (row stride 3072 halves) into LDS with
// 4-DWORD padding after every row (32 DWORDs) -> LDS pitch 72 halves.
// D# per CDNA5 ISA 8.3/8.4: group0 = {count|flags, lds_addr, global_addr, type=2},
// group1 = {data_size/pad, dims, tile dims, strides}.  (TENSORcnt tracked.)
__device__ __forceinline__ void tdm_load_tile(unsigned lds_addr, const _Float16* gptr)
{
    const unsigned long long ga = (unsigned long long)(uintptr_t)gptr;
    u32x4 g0 = { 1u,                                   // count=1, user mode
                 lds_addr,                             // LDS byte address
                 (unsigned)ga,                         // global_addr[31:0]
                 (unsigned)((ga >> 32) & 0x01FFFFFFull) | (2u << 30) }; // [56:32] | type=2
    i32x8 g1 = { (1 << 16) | (1 << 20) | (4 << 22) | (3 << 25),
                 //  data_size=2B | pad_enable | pad_interval=32dw | pad_amount=4dw
                 (64 << 16),          // tensor_dim0 = 64 (lo16 at bits[31:16])
                 (2048 << 16),        // tensor_dim0 hi=0 | tensor_dim1 lo=2048
                 (64 << 16),          // tensor_dim1 hi=0 | tile_dim0 = 64
                 32,                  // tile_dim1 = 32 | tile_dim2 = 0
                 3072,                // tensor_dim0_stride lo32 (halves)
                 0, 0 };              // stride0 hi | tensor_dim1_stride = 0
    const i32x4 z4 = { 0, 0, 0, 0 };              // groups 2/3 unused (2D)
    const i32x8 z8 = { 0, 0, 0, 0, 0, 0, 0, 0 };  // extra group (clang-23 form)
    __builtin_amdgcn_tensor_load_to_lds(g0, g1, z4, z4, z8, 0);
}

// ---------------------------------------------------------------------------
// Generic tiled GEMM with bias:  C[M,N] = A[M,K] * B[K,N] + bias[N]
// Block tile 128x128, K-step 32, 8 waves; each wave 32x64 = 2x4 WMMA tiles.
// ---------------------------------------------------------------------------
template <typename TA, typename TC>
__global__ __launch_bounds__(256)
void gemm_bias_kernel(const TA* __restrict__ A, const float* __restrict__ B,
                      const float* __restrict__ bias, TC* __restrict__ C,
                      int M, int N, int K)
{
    constexpr int BM = 128, BN = 128, BK = 32, PAD = 40; // 80B pitch (16B mult)
    __shared__ __align__(16) _Float16 As[BM * PAD];      // [m][k]
    __shared__ __align__(16) _Float16 Bs[BN * PAD];      // transposed [n][k]

    const int tid  = threadIdx.x;
    const int lane = tid & 31;
    const int wave = tid >> 5;
    const int wy   = wave >> 1;       // 0..3 : 32-row band
    const int wx   = wave & 1;        // 0..1 : 64-col band
    const int bm0  = blockIdx.y * BM;
    const int bn0  = blockIdx.x * BN;

    const int m16  = lane & 15;
    const int kb_a = (lane < 16) ? 0 : 16;  // A-fragment K base (contiguous k = 16h + s)
    const int kb_b = (lane < 16) ? 0 : 16;  // B-fragment K base

    f32x8 acc[2][4] = {};

    for (int kb = 0; kb < K; kb += BK) {
        __syncthreads();
        // ---- stage A tile (BM x BK) -> f16 LDS -------------------------
        {
            const int m0 = tid >> 3;          // 0..31
            const int kc = (tid & 7) * 4;     // 0..28
            #pragma unroll
            for (int p = 0; p < 4; ++p) {
                const int m = m0 + p * 32;
                const TA* src = A + (size_t)(bm0 + m) * K + kb + kc;
                if (kb + BK < K) __builtin_prefetch(src + BK, 0, 1);
                #pragma unroll
                for (int i = 0; i < 4; ++i)
                    As[m * PAD + kc + i] = (_Float16)src[i];
            }
        }
        // ---- stage B tile (BK x BN) transposed -> f16 LDS --------------
        {
            const int n  = tid & 127;
            const int k0 = tid >> 7;          // 0..1
            #pragma unroll
            for (int p = 0; p < 16; ++p) {
                const int k = k0 + p * 2;
                Bs[n * PAD + k] = (_Float16)B[(size_t)(kb + k) * N + bn0 + n];
            }
        }
        __syncthreads();

        // ---- fragment loads (16B-aligned ds_load_b128 chunks) ----------
        h16x16 af[2], bf[4];
        #pragma unroll
        for (int i = 0; i < 2; ++i) {
            const _Float16* ap = &As[(wy * 32 + i * 16 + m16) * PAD + kb_a];
            h16x8 lo = *(const h16x8*)ap;
            h16x8 hi = *(const h16x8*)(ap + 8);
            #pragma unroll
            for (int e = 0; e < 8; ++e) { af[i][e] = lo[e]; af[i][8 + e] = hi[e]; }
        }
        #pragma unroll
        for (int j = 0; j < 4; ++j) {
            const _Float16* bp = &Bs[(wx * 64 + j * 16 + m16) * PAD + kb_b];
            h16x8 lo = *(const h16x8*)bp;
            h16x8 hi = *(const h16x8*)(bp + 8);
            #pragma unroll
            for (int e = 0; e < 8; ++e) { bf[j][e] = lo[e]; bf[j][8 + e] = hi[e]; }
        }
        #pragma unroll
        for (int i = 0; i < 2; ++i)
            #pragma unroll
            for (int j = 0; j < 4; ++j)
                acc[i][j] = wmma_f16(af[i], bf[j], acc[i][j]);
    }

    // ---- epilogue: bias add, store per C/D VGPR layout -----------------
    const int mb  = (lane < 16) ? 0 : 8;
    const int n16 = lane & 15;
    #pragma unroll
    for (int i = 0; i < 2; ++i)
        #pragma unroll
        for (int j = 0; j < 4; ++j) {
            const int col = bn0 + wx * 64 + j * 16 + n16;
            const float bv = bias[col];
            #pragma unroll
            for (int r = 0; r < 8; ++r) {
                const int row = bm0 + wy * 32 + i * 16 + mb + r;
                C[(size_t)row * N + col] = (TC)(acc[i][j][r] + bv);
            }
        }
}

// ---------------------------------------------------------------------------
// Flash attention: qkv f16 [B, N, 3, H, 64] -> out f16 [B, N, H*64].
// Block = (b,h, 128 query rows); 8 waves x 16 rows each; key step 32.
// K/V tiles staged into LDS by the Tensor Data Mover (wave 0 issues).
// ---------------------------------------------------------------------------
__global__ __launch_bounds__(256)
void attn_kernel(const _Float16* __restrict__ qkv, _Float16* __restrict__ outh)
{
    constexpr int SQ = 3 * DIM;   // qkv row stride (halves)
    constexpr int KP = 72;        // K/V LDS pitch (64 + TDM pad; 144B, 16B mult)
    constexpr int PP = 48;        // P  LDS pitch  (96B, 16B mult)
    __shared__ __align__(16) _Float16 Ks[32 * KP];
    __shared__ __align__(16) _Float16 Vs[32 * KP];
    __shared__ __align__(16) _Float16 Ps[8 * 16 * PP];

    const int tid  = threadIdx.x;
    const int lane = tid & 31;
    const int wave = tid >> 5;
    const int b  = blockIdx.x >> 4;
    const int h  = blockIdx.x & 15;
    const int n0 = blockIdx.y * 128 + wave * 16;

    const int m16  = lane & 15;
    const int kb_a = (lane < 16) ? 0 : 16;
    const int kb_b = (lane < 16) ? 0 : 16;
    const int mb   = (lane < 16) ? 0 : 8;

    const unsigned ldsK = (unsigned)(uintptr_t)&Ks[0];
    const unsigned ldsV = (unsigned)(uintptr_t)&Vs[0];

    // Q fragments for d=0..31 / 32..63, pre-scaled by 1/sqrt(64) = 0.125
    h16x16 q[2];
    {
        const _Float16* qr = qkv + (size_t)(b * SEQ + n0 + m16) * SQ + h * HDIM;
        #pragma unroll
        for (int c = 0; c < 2; ++c)
            #pragma unroll
            for (int e = 0; e < 8; ++e) {
                q[c][e]     = qr[c * 32 + kb_a + e]      * (_Float16)0.125f;
                q[c][8 + e] = qr[c * 32 + kb_a + 8 + e] * (_Float16)0.125f;
            }
    }

    f32x8 o[4] = {};
    float mrow[8], lrow[8];
    #pragma unroll
    for (int r = 0; r < 8; ++r) { mrow[r] = -1e30f; lrow[r] = 0.f; }

    _Float16* Pw = &Ps[wave * 16 * PP];

    for (int j = 0; j < SEQ; j += 32) {
        __syncthreads();                       // all waves done with prev tiles
        if (wave == 0) {
            const _Float16* kp = qkv + (size_t)(b * SEQ + j) * SQ + DIM     + h * HDIM;
            const _Float16* vp = qkv + (size_t)(b * SEQ + j) * SQ + 2 * DIM + h * HDIM;
            tdm_load_tile(ldsK, kp);           // tensor_load_to_lds (TDM)
            tdm_load_tile(ldsV, vp);
            __builtin_amdgcn_s_wait_tensorcnt(0);
        }
        __syncthreads();                       // tiles visible to all waves

        // ---- S = (Q*scale) K^T : 2 key-subtiles x 2 d-chunks -----------
        f32x8 s[2] = {};
        #pragma unroll
        for (int t = 0; t < 2; ++t)
            #pragma unroll
            for (int c = 0; c < 2; ++c) {
                const _Float16* kp = &Ks[(t * 16 + m16) * KP + c * 32 + kb_b];
                h16x8 lo = *(const h16x8*)kp;
                h16x8 hi = *(const h16x8*)(kp + 8);
                h16x16 kf;
                #pragma unroll
                for (int e = 0; e < 8; ++e) { kf[e] = lo[e]; kf[8 + e] = hi[e]; }
                s[t] = wmma_f16(q[c], kf, s[t]);
            }

        // ---- online softmax (row reductions across 16-lane halves) -----
        float p0[8], p1[8], alpha[8];
        #pragma unroll
        for (int r = 0; r < 8; ++r) {
            float mx = fmaxf(s[0][r], s[1][r]);
            #pragma unroll
            for (int d = 1; d < 16; d <<= 1)
                mx = fmaxf(mx, __shfl_xor(mx, d, 16));
            const float mn = fmaxf(mrow[r], mx);
            alpha[r] = __expf(mrow[r] - mn);
            mrow[r]  = mn;
            p0[r] = __expf(s[0][r] - mn);
            p1[r] = __expf(s[1][r] - mn);
            float rs = p0[r] + p1[r];
            #pragma unroll
            for (int d = 1; d < 16; d <<= 1)
                rs += __shfl_xor(rs, d, 16);
            lrow[r] = lrow[r] * alpha[r] + rs;
        }
        #pragma unroll
        for (int t = 0; t < 4; ++t)
            #pragma unroll
            for (int r = 0; r < 8; ++r)
                o[t][r] *= alpha[r];

        // ---- P: C-layout -> A-layout via per-wave LDS bounce -----------
        #pragma unroll
        for (int r = 0; r < 8; ++r) {
            Pw[(mb + r) * PP + m16]      = (_Float16)p0[r];
            Pw[(mb + r) * PP + 16 + m16] = (_Float16)p1[r];
        }
        asm volatile("s_wait_dscnt 0" ::: "memory");  // wave-local LDS RAW
        h16x16 pf;
        {
            const _Float16* pp = &Pw[m16 * PP + kb_a];
            h16x8 lo = *(const h16x8*)pp;
            h16x8 hi = *(const h16x8*)(pp + 8);
            #pragma unroll
            for (int e = 0; e < 8; ++e) { pf[e] = lo[e]; pf[8 + e] = hi[e]; }
        }

        // ---- O += P V : 4 d-subtiles -----------------------------------
        #pragma unroll
        for (int t = 0; t < 4; ++t) {
            h16x16 vf;
            #pragma unroll
            for (int e = 0; e < 16; ++e)
                vf[e] = Vs[(kb_b + e) * KP + t * 16 + m16];
            o[t] = wmma_f16(pf, vf, o[t]);
        }
    }

    // ---- normalize & store [B, N, C] f16 -------------------------------
    float inv[8];
    #pragma unroll
    for (int r = 0; r < 8; ++r) inv[r] = 1.f / lrow[r];
    #pragma unroll
    for (int t = 0; t < 4; ++t)
        #pragma unroll
        for (int r = 0; r < 8; ++r) {
            const int row = n0 + mb + r;
            outh[(size_t)(b * SEQ + row) * DIM + h * HDIM + t * 16 + m16] =
                (_Float16)(o[t][r] * inv[r]);
        }
}

// ---------------------------------------------------------------------------
extern "C" void kernel_launch(void* const* d_in, const int* in_sizes, int n_in,
                              void* d_out, int out_size, void* d_ws, size_t ws_size,
                              hipStream_t stream)
{
    (void)in_sizes; (void)n_in; (void)out_size; (void)ws_size;
    const float* x      = (const float*)d_in[0];
    const float* W_qkv  = (const float*)d_in[1];
    const float* b_qkv  = (const float*)d_in[2];
    const float* W_proj = (const float*)d_in[3];
    const float* b_proj = (const float*)d_in[4];
    float*       out    = (float*)d_out;

    // workspace: qkv f16 (25.2MB) | attn f16 (8.4MB)
    _Float16* qkv_h  = (_Float16*)d_ws;
    _Float16* attn_h = qkv_h + (size_t)BATCH * SEQ * 3 * DIM;

    const int M = BATCH * SEQ;                  // 4096

    // 1) qkv = x @ W_qkv + b_qkv        (fp32 in -> f16 out)
    gemm_bias_kernel<float, _Float16>
        <<<dim3((3 * DIM) / 128, M / 128), 256, 0, stream>>>(
            x, W_qkv, b_qkv, qkv_h, M, 3 * DIM, DIM);

    // 2) flash attention over all heads (TDM-staged K/V tiles)
    attn_kernel<<<dim3(BATCH * NHEAD, SEQ / 128), 256, 0, stream>>>(qkv_h, attn_h);

    // 3) out = attn @ W_proj + b_proj   (f16 in -> fp32 out)
    gemm_bias_kernel<_Float16, float>
        <<<dim3(DIM / 128, M / 128), 256, 0, stream>>>(
            attn_h, W_proj, b_proj, out, M, DIM, DIM);
}
